// OptimizedCmxuLayer_26723286516361
// MI455X (gfx1250) — matrix operand order repaired and checked by reading, not compiled, not run on hardware
//
#include <hip/hip_runtime.h>

// ---------------------------------------------------------------------------
// Clements mesh (N=128) applied to 65536 complex fields on MI455X (gfx1250).
//   Kernel A: build U (LDS-resident, layer-serial), embed complex matmul as a
//             real 256x256 matrix W, emit f16 hi/lo WMMA B-fragments to d_ws.
//   Kernel B: out(65536x256) = [Fr|Fi] @ W via v_wmma_f32_16x16x32_f16 with
//             3-term hi/lo split for ~f32 accuracy. Memory floor ~5.5us.
// ---------------------------------------------------------------------------

typedef __attribute__((ext_vector_type(16))) _Float16 v16h;
typedef __attribute__((ext_vector_type(8)))  float    v8f;

#define WMMA_F16(A, B, C) \
  __builtin_amdgcn_wmma_f32_16x16x32_f16(false, (A), false, (B), (short)0, (C), false, false)

__device__ __forceinline__ int pack_h2(_Float16 a, _Float16 b) {
  union { _Float16 h[2]; int i; } u;
  u.h[0] = a; u.h[1] = b;           // little-endian: h[0] -> dword[15:0] = lower K
  return u.i;
}

// ---------------------------------------------------------------------------
// Kernel A: build W fragments. One workgroup (256 threads, 8 waves).
// LDS: Ur[128*128] f32, Ui[128*128] f32, 5*64 coeffs -> ~132KB dynamic LDS.
// d_ws layout (ints): whi[32768] then wlo[32768]  (256 KB total).
// Fragment dword address: ((kb*16 + nb)*32 + lane)*8 + v
//   B-operand layout (16-bit, 32Kx16N): lane<16 -> K=0..7,16..23; lane>=16 -> K=8..15,24..31
//   n = nb*16 + (lane&15); dword v packs K = kb*32 + 8*(lane>>4) + 16*(v>>2) + 2*(v&3) (+t in hi half)
// ---------------------------------------------------------------------------
__global__ void __launch_bounds__(256) cmxu_build_w(const float* __restrict__ mzi,
                                                    const float* __restrict__ oph,
                                                    int* __restrict__ wfrag) {
  extern __shared__ char smemA[];
  float* Ur = (float*)smemA;        // 16384
  float* Ui = Ur + 16384;           // 16384
  float* Ar = Ui + 16384;           // 64
  float* Ai = Ar + 64;
  float* Sv = Ar + 128;
  float* Dr = Ar + 192;
  float* Di = Ar + 256;

  const int tid = threadIdx.x;

  // U = I
  for (int e = tid; e < 16384; e += 256) {
    Ur[e] = ((e >> 7) == (e & 127)) ? 1.0f : 0.0f;
    Ui[e] = 0.0f;
  }
  __syncthreads();

  // 128 layers of disjoint 2x2 Givens column rotations
  int idx = 0;
  for (int layer = 0; layer < 128; ++layer) {
    const int p = layer & 1;
    const int m = p ? 63 : 64;
    if (tid < m) {
      const float th = mzi[idx + 2 * tid];
      const float ph = mzi[idx + 2 * tid + 1];
      float s, c, sp, cp;
      __sincosf(th, &s, &c);
      __sincosf(ph, &sp, &cp);
      Ar[tid] = c * cp;   // a = c*e^{i phi}
      Ai[tid] = c * sp;
      Sv[tid] = s;        // s (real)
      Dr[tid] = -c * cp;  // d = -c*conj(e^{i phi})
      Di[tid] = c * sp;
    }
    idx += 2 * m;
    __syncthreads();

    const int tot = m << 7;  // m pairs * 128 rows
    for (int e = tid; e < tot; e += 256) {
      const int j = e >> 7, r = e & 127;
      const int k = p + 2 * j;
      float* pr = Ur + r * 128 + k;
      float* pi = Ui + r * 128 + k;
      const float cir = pr[0], cjr = pr[1];
      const float cii = pi[0], cji = pi[1];
      const float ar = Ar[j], ai = Ai[j], s = Sv[j], dr = Dr[j], di = Di[j];
      pr[0] = cir * ar - cii * ai + cjr * s;   // U[:,k]   = ci*a + cj*s
      pi[0] = cir * ai + cii * ar + cji * s;
      pr[1] = cir * s + cjr * dr - cji * di;   // U[:,k+1] = ci*s + cj*d
      pi[1] = cii * s + cjr * di + cji * dr;
    }
    __syncthreads();
  }

  // column output phases: U[:,c] *= e^{i oph[c]}
  for (int e = tid; e < 16384; e += 256) {
    const int c = e & 127;
    float sp, cp;
    __sincosf(oph[c], &sp, &cp);
    const float ur = Ur[e], ui = Ui[e];
    Ur[e] = ur * cp - ui * sp;
    Ui[e] = ur * sp + ui * cp;
  }
  __syncthreads();

  // Emit W = [[Ur, Ui], [-Ui, Ur]] with interleaved output columns,
  // split into f16 hi/lo, laid out as per-lane WMMA B fragments.
  int* whi = wfrag;
  int* wlo = wfrag + 32768;
  for (int dw = tid; dw < 32768; dw += 256) {
    const int v    = dw & 7;
    const int lane = (dw >> 3) & 31;
    const int f    = dw >> 8;
    const int nb   = f & 15;
    const int kb   = f >> 4;
    const int n    = nb * 16 + (lane & 15);
    const int k    = kb * 32 + ((lane >> 4) << 3) + ((v & 4) << 2) + (v & 3) * 2;
    const int nn   = n >> 1;
    float w[2];
#pragma unroll
    for (int t = 0; t < 2; ++t) {
      const int kk = k + t;
      if (kk < 128) w[t] = (n & 1) ? Ui[kk * 128 + nn] : Ur[kk * 128 + nn];
      else          w[t] = (n & 1) ? Ur[(kk - 128) * 128 + nn] : -Ui[(kk - 128) * 128 + nn];
    }
    const _Float16 h0 = (_Float16)w[0], h1 = (_Float16)w[1];
    const _Float16 l0 = (_Float16)(w[0] - (float)h0);
    const _Float16 l1 = (_Float16)(w[1] - (float)h1);
    whi[dw] = pack_h2(h0, h1);
    wlo[dw] = pack_h2(l0, l1);
  }
}

// ---------------------------------------------------------------------------
// Kernel B: out = [Fr|Fi] @ W.  512 blocks x 256 threads (8 waves).
// Each wave: 16 M-rows x full N=256 -> 16 accumulator tiles (128 VGPRs).
// LDS: full Whi+Wlo fragment image (256 KB of the 320 KB WGP budget).
// Per output tile: 3 x v_wmma_f32_16x16x32_f16 (hi*hi + hi*lo + lo*hi).
// ---------------------------------------------------------------------------
__global__ void __launch_bounds__(256) cmxu_gemm(const float* __restrict__ fr,
                                                 const float* __restrict__ fi,
                                                 const int* __restrict__ wfrag,
                                                 float* __restrict__ out) {
  extern __shared__ char smemB[];
  int* lds = (int*)smemB;

  // Stage 256 KB of W fragments into LDS (coalesced b128 copies).
  {
    const int4* src = (const int4*)wfrag;
    int4* dst = (int4*)lds;
    for (int i = threadIdx.x; i < 16384; i += 256) dst[i] = src[i];
  }
  __syncthreads();

  const int lane  = threadIdx.x & 31;
  const int wave  = threadIdx.x >> 5;
  const int khalf = lane >> 4;       // K-half selector (A layout)
  const int mrow  = lane & 15;       // M within tile (A/D layout), N within tile (B/D)

  const long rowA = (long)blockIdx.x * 128 + wave * 16 + mrow;
  const float2* frowR = (const float2*)(fr + rowA * 128);
  const float2* frowI = (const float2*)(fi + rowA * 128);

  const v8f vzero = {0.f, 0.f, 0.f, 0.f, 0.f, 0.f, 0.f, 0.f};
  v8f acc[16];
#pragma unroll
  for (int i = 0; i < 16; ++i) acc[i] = vzero;

  for (int kb = 0; kb < 8; ++kb) {
    const float2* row2 = (kb < 4) ? frowR : frowI;
    const int kbase = (kb & 3) * 32 + khalf * 8;

    // A fragment, hi/lo split, per documented 16-bit 16x32 A layout:
    // lane<16: v0..3 -> K=0..7, v4..7 -> K=16..23 ; lane>=16: +8 / +24
    union { int i[8]; v16h h; } ahi, alo;
#pragma unroll
    for (int v = 0; v < 8; ++v) {
      const int k = kbase + ((v & 4) << 2) + (v & 3) * 2;
      const float2 f2 = row2[k >> 1];
      const _Float16 hx = (_Float16)f2.x, hy = (_Float16)f2.y;
      const _Float16 lx = (_Float16)(f2.x - (float)hx);
      const _Float16 ly = (_Float16)(f2.y - (float)hy);
      ahi.i[v] = pack_h2(hx, hy);
      alo.i[v] = pack_h2(lx, ly);
    }

#pragma unroll
    for (int nb = 0; nb < 16; ++nb) {
      const int base = ((kb * 16 + nb) * 32 + lane) * 8;  // 32B-aligned per lane
      union { int4 q[2]; v16h h; } bhi, blo;
      const int4* ph = (const int4*)(lds + base);
      const int4* pl = (const int4*)(lds + 32768 + base);
      bhi.q[0] = ph[0]; bhi.q[1] = ph[1];
      blo.q[0] = pl[0]; blo.q[1] = pl[1];
      acc[nb] = WMMA_F16(ahi.h, bhi.h, acc[nb]);  // hi * hi
      acc[nb] = WMMA_F16(ahi.h, blo.h, acc[nb]);  // hi * lo
      acc[nb] = WMMA_F16(alo.h, bhi.h, acc[nb]);  // lo * hi
    }
  }

  // D tile store (16x16 f32 layout): VGPR r -> M = r (+8 for upper lanes),
  // N = lane&15. Half-wave rows are 64B-contiguous in the interleaved output.
  const long rowD = (long)blockIdx.x * 128 + wave * 16 + khalf * 8;
#pragma unroll
  for (int nb = 0; nb < 16; ++nb) {
    const int col = nb * 16 + mrow;
#pragma unroll
    for (int r = 0; r < 8; ++r) {
      out[(rowD + r) * 256 + col] = acc[nb][r];
    }
  }
}

// ---------------------------------------------------------------------------
extern "C" void kernel_launch(void* const* d_in, const int* in_sizes, int n_in,
                              void* d_out, int out_size, void* d_ws, size_t ws_size,
                              hipStream_t stream) {
  const float* fr  = (const float*)d_in[0];   // field_real  65536x128
  const float* fi  = (const float*)d_in[1];   // field_imag  65536x128
  const float* mzi = (const float*)d_in[2];   // 16256 phases
  const float* oph = (const float*)d_in[3];   // 128 output phases
  float* out = (float*)d_out;                 // 65536x128 complex64 (interleaved)
  int* wfrag = (int*)d_ws;                    // 256 KB scratch: whi | wlo

  const size_t ldsA = (size_t)(32768 + 320) * sizeof(float);  // ~132 KB
  const size_t ldsB = (size_t)262144;                         // 256 KB

  // Allow >64KB dynamic LDS (non-stream call, safe under graph capture).
  (void)hipFuncSetAttribute((const void*)cmxu_build_w,
                            hipFuncAttributeMaxDynamicSharedMemorySize, (int)ldsA);
  (void)hipFuncSetAttribute((const void*)cmxu_gemm,
                            hipFuncAttributeMaxDynamicSharedMemorySize, (int)ldsB);

  cmxu_build_w<<<1, 256, ldsA, stream>>>(mzi, oph, wfrag);
  cmxu_gemm<<<512, 256, ldsB, stream>>>(fr, fi, (const int*)wfrag, out);
}
